// AttentionHead_6227702579667
// MI455X (gfx1250) — compile-verified, matrix operand order checked by reading, not run on hardware
//
#include <hip/hip_runtime.h>
#include <hip/hip_bf16.h>

#define BB 8
#define TT 2048
#define DD 512
#define HH 64

typedef __attribute__((ext_vector_type(16))) __bf16 v16bf;
typedef __attribute__((ext_vector_type(8)))  float  v8f;
typedef __attribute__((__vector_size__(16))) int    vi4;   // int4 vector for async builtin

#define AS1 __attribute__((address_space(1)))
#define AS3 __attribute__((address_space(3)))

#if __has_builtin(__builtin_amdgcn_global_load_async_to_lds_b128) && \
    __has_builtin(__builtin_amdgcn_s_wait_asynccnt)
#define ASYNC_STAGE 1
#else
#define ASYNC_STAGE 0
#endif

__device__ __forceinline__ unsigned short f2bf(float f) {
  unsigned int u = __float_as_uint(f);
  unsigned int r = (u + 0x7FFFu + ((u >> 16) & 1u)) >> 16;  // RNE
  return (unsigned short)r;
}

// Build a 16-element bf16 WMMA fragment from two 16-byte chunks
// (8 K-values at p0, 8 K-values at p0+32B == +16 K).
__device__ __forceinline__ v16bf ldfrag(const void* p0, const void* p1) {
  union { v16bf v; uint4 q[2]; } u;
  u.q[0] = *(const uint4*)p0;
  u.q[1] = *(const uint4*)p1;
  return u.v;
}

#if ASYNC_STAGE
__device__ __forceinline__ void async_cp16(const void* g, void* l) {
  // per-lane 16B global -> LDS DMA, tracked by ASYNCcnt
  __builtin_amdgcn_global_load_async_to_lds_b128((AS1 vi4*)g, (AS3 vi4*)l, 0, 0);
}
#endif

// ---------------- conversion kernels ----------------

__global__ void cvt_x(const float* __restrict__ x, unsigned short* __restrict__ xb, int n4) {
  int i = blockIdx.x * blockDim.x + threadIdx.x;
  if (i >= n4) return;
  float4 f = ((const float4*)x)[i];
  union { unsigned short s[4]; uint2 u; } o;
  o.s[0] = f2bf(f.x); o.s[1] = f2bf(f.y); o.s[2] = f2bf(f.z); o.s[3] = f2bf(f.w);
  ((uint2*)xb)[i] = o.u;
}

// W [D,H] fp32 -> Wt [mat][H][D] bf16 (transposed so K-pairs are contiguous)
__global__ void cvt_w(const float* __restrict__ Wq, const float* __restrict__ Wk,
                      const float* __restrict__ Wv, unsigned short* __restrict__ wt) {
  int i = blockIdx.x * blockDim.x + threadIdx.x;
  if (i >= 3 * DD * HH) return;
  int mat = i / (DD * HH);
  int rem = i % (DD * HH);
  int k = rem / HH, n = rem % HH;
  const float* W = (mat == 0) ? Wq : ((mat == 1) ? Wk : Wv);
  wt[(size_t)mat * HH * DD + (size_t)n * DD + k] = f2bf(W[(size_t)k * HH + n]);
}

// ---------------- projection GEMM: Q,K,V = x@W + b ----------------
// One wave computes a 16x64 output tile of one matrix.
// Q is pre-scaled by (1/sqrt(64))*log2(e) so flash softmax runs in base 2.
// V is stored transposed [B][H][T].

__global__ __launch_bounds__(32) void proj(const unsigned short* __restrict__ xb,
                                           const unsigned short* __restrict__ wt,
                                           const float* __restrict__ bq,
                                           const float* __restrict__ bk,
                                           const float* __restrict__ bv,
                                           unsigned short* __restrict__ Qb,
                                           unsigned short* __restrict__ Kb,
                                           unsigned short* __restrict__ Vt) {
  const int tile = blockIdx.x;      // 0 .. B*T/16-1
  const int mat  = blockIdx.y;      // 0=Q 1=K 2=V
  const int lane = threadIdx.x;
  const int ln = lane & 15, hg = lane >> 4;
  const int row0 = tile * 16;
  const unsigned short* W = wt + (size_t)mat * HH * DD;
  const float* bias = (mat == 0) ? bq : ((mat == 1) ? bk : bv);

  v8f acc[4];
  for (int t = 0; t < 4; ++t)
    for (int r = 0; r < 8; ++r) acc[t][r] = 0.f;

  for (int kk = 0; kk < DD / 32; ++kk) {
    const char* ap = (const char*)xb + (size_t)(row0 + ln) * DD * 2 + kk * 64 + hg * 16;
    v16bf a = ldfrag(ap, ap + 32);
    for (int t = 0; t < 4; ++t) {
      const char* bp = (const char*)W + (size_t)(t * 16 + ln) * DD * 2 + kk * 64 + hg * 16;
      v16bf bf = ldfrag(bp, bp + 32);
      acc[t] = __builtin_amdgcn_wmma_f32_16x16x32_bf16(false, a, false, bf,
                                                       (short)0, acc[t], false, false);
    }
  }

  // fold 1/sqrt(H) AND log2(e) into Q
  const float scale = (mat == 0) ? (0.125f * 1.44269504088896340736f) : 1.0f;
  for (int t = 0; t < 4; ++t) {
    const int n = t * 16 + ln;
    const float bb = bias[n];
    for (int r = 0; r < 8; ++r) {
      const int m = row0 + r + 8 * hg;
      unsigned short h = f2bf((acc[t][r] + bb) * scale);
      if (mat == 0)      Qb[(size_t)m * HH + n] = h;
      else if (mat == 1) Kb[(size_t)m * HH + n] = h;
      else {
        int bidx = m / TT, rb = m % TT;
        Vt[((size_t)bidx * HH + n) * TT + rb] = h;
      }
    }
  }
}

// ---------------- flash attention ----------------
// 4 waves / block, each wave owns 16 query rows; key tiles of 64.
// All 4 waves share the same key-tile count (nkt = qt+1), so K/V tiles are
// staged cooperatively into double-buffered LDS with async-to-LDS DMA.

__global__ __launch_bounds__(128) void flash(const unsigned short* __restrict__ Qb,
                                             const unsigned short* __restrict__ Kb,
                                             const unsigned short* __restrict__ Vt,
                                             const unsigned char* __restrict__ pad,
                                             float* __restrict__ out) {
#if ASYNC_STAGE
  __shared__ unsigned short sK[2][64 * HH];   // [key][h]  8KB each buffer
  __shared__ unsigned short sV[2][64 * HH];   // [h][key]  8KB each buffer
#endif
  __shared__ unsigned short sP[4][16 * HH];   // per-wave P tile (C-layout -> A-layout)

  const int w    = threadIdx.x >> 5;
  const int lane = threadIdx.x & 31;
  const int ln = lane & 15, hg = lane >> 4;
  const int b  = blockIdx.x >> 5;             // 32 query tiles per batch
  const int qt = blockIdx.x & 31;
  const int q0 = qt * 64 + w * 16;            // first query row (in batch) for this wave
  const size_t rowbase = (size_t)b * TT;

  const unsigned short* Kbb = Kb + rowbase * HH;
  const unsigned short* Vbb = Vt + (size_t)b * HH * TT;
  const int nkt = qt + 1;                     // identical for all 4 waves

  // Q fragments for both K-steps of the H=64 reduction
  v16bf qf[2];
  for (int kk = 0; kk < 2; ++kk) {
    const char* p = (const char*)Qb + (rowbase + q0 + ln) * HH * 2 + kk * 64 + hg * 16;
    qf[kk] = ldfrag(p, p + 32);
  }

  v8f o[4];
  float mi[8], li[8];
  for (int t = 0; t < 4; ++t)
    for (int r = 0; r < 8; ++r) o[t][r] = 0.f;
  for (int r = 0; r < 8; ++r) { mi[r] = -3.0e38f; li[r] = 0.f; }

#if ASYNC_STAGE
  // issue DMA for tile 0 (K: 512 16B chunks, V: 512 16B chunks, split over 128 threads)
  {
    const int tid = threadIdx.x;
    for (int j = 0; j < 4; ++j) {
      const int c = j * 128 + tid;
      async_cp16((const char*)Kbb + (size_t)c * 16, (char*)&sK[0][0] + c * 16);
    }
    for (int j = 0; j < 4; ++j) {
      const int c = j * 128 + tid;
      const int h = c >> 3, x16 = c & 7;
      async_cp16((const char*)Vbb + (size_t)h * TT * 2 + x16 * 16, (char*)&sV[0][0] + c * 16);
    }
  }
#endif

  for (int kt = 0; kt < nkt; ++kt) {
    const int kb = kt * 64;
    const bool diag = (kt == nkt - 1);   // only the last tile crosses the causal diagonal
#if ASYNC_STAGE
    const int cur = kt & 1;
    __builtin_amdgcn_s_wait_asynccnt(0);   // my chunks of tile kt have landed in LDS
    __syncthreads();                       // everyone's chunks landed; prev compute retired
    if (kt + 1 < nkt) {                    // overlap tile kt+1 DMA with tile kt compute
      const int tid = threadIdx.x;
      const size_t kb2 = (size_t)(kb + 64);
      for (int j = 0; j < 4; ++j) {
        const int c = j * 128 + tid;
        async_cp16((const char*)Kbb + kb2 * 128 + (size_t)c * 16,
                   (char*)&sK[cur ^ 1][0] + c * 16);
      }
      for (int j = 0; j < 4; ++j) {
        const int c = j * 128 + tid;
        const int h = c >> 3, x16 = c & 7;
        async_cp16((const char*)Vbb + (size_t)h * TT * 2 + kb2 * 2 + x16 * 16,
                   (char*)&sV[cur ^ 1][0] + c * 16);
      }
    }
    const char* Kt = (const char*)&sK[cur][0];
    const char* Vtile = (const char*)&sV[cur][0];
#else
    if (kt + 1 < nkt) {
      __builtin_prefetch((const char*)Kbb + (size_t)(kb + 64 + 2 * lane) * HH * 2, 0, 0);
      __builtin_prefetch((const char*)Vbb + (size_t)(2 * lane) * TT * 2 + (kb + 64) * 2, 0, 0);
    }
#endif

    // S = Q @ K^T  (Q pre-scaled by 0.125*log2e)
    v8f s[4];
    for (int t = 0; t < 4; ++t)
      for (int r = 0; r < 8; ++r) s[t][r] = 0.f;
    for (int t = 0; t < 4; ++t) {
      for (int kk = 0; kk < 2; ++kk) {
#if ASYNC_STAGE
        const char* p = Kt + (size_t)(t * 16 + ln) * 128 + kk * 64 + hg * 16;
#else
        const char* p = (const char*)Kbb + (size_t)(kb + t * 16 + ln) * HH * 2 + kk * 64 + hg * 16;
#endif
        v16bf kf = ldfrag(p, p + 32);
        s[t] = __builtin_amdgcn_wmma_f32_16x16x32_bf16(false, qf[kk], false, kf,
                                                       (short)0, s[t], false, false);
      }
    }

    // masks + new row max (causal compare only on the diagonal tile)
    float mnew[8];
    for (int r = 0; r < 8; ++r) mnew[r] = mi[r];
    for (int t = 0; t < 4; ++t) {
      const int kg = kb + t * 16 + ln;
      const bool pm = pad[(size_t)b * TT + kg] != 0;
      if (diag) {
        for (int r = 0; r < 8; ++r) {
          const int qg = q0 + r + 8 * hg;
          float sv = s[t][r];
          if (pm || (kg > qg)) sv = -1.0e30f;
          s[t][r] = sv;
          mnew[r] = fmaxf(mnew[r], sv);
        }
      } else {
        for (int r = 0; r < 8; ++r) {
          float sv = pm ? -1.0e30f : s[t][r];
          s[t][r] = sv;
          mnew[r] = fmaxf(mnew[r], sv);
        }
      }
    }
    // row max across the 16 lanes of each half-wave
    for (int r = 0; r < 8; ++r) {
      float v = mnew[r];
      v = fmaxf(v, __shfl_xor(v, 1));
      v = fmaxf(v, __shfl_xor(v, 2));
      v = fmaxf(v, __shfl_xor(v, 4));
      v = fmaxf(v, __shfl_xor(v, 8));
      mnew[r] = v;
    }

    // p = 2^(s - mnew), row sums
    float rs[8];
    for (int r = 0; r < 8; ++r) rs[r] = 0.f;
    for (int t = 0; t < 4; ++t)
      for (int r = 0; r < 8; ++r) {
        float pv = exp2f(s[t][r] - mnew[r]);
        s[t][r] = pv;
        rs[r] += pv;
      }
    for (int r = 0; r < 8; ++r) {
      float v = rs[r];
      v += __shfl_xor(v, 1);
      v += __shfl_xor(v, 2);
      v += __shfl_xor(v, 4);
      v += __shfl_xor(v, 8);
      float alpha = exp2f(mi[r] - mnew[r]);
      li[r] = li[r] * alpha + v;
      mi[r] = mnew[r];
      for (int t = 0; t < 4; ++t) o[t][r] *= alpha;
    }

    // P (C-layout, fp32) -> LDS bf16 [16 rows][64 keys]; wave-private, DS in-order
    unsigned short* Pl = &sP[w][0];
    for (int t = 0; t < 4; ++t) {
      const int key = t * 16 + ln;
      for (int r = 0; r < 8; ++r) {
        const int m = r + 8 * hg;
        Pl[m * HH + key] = f2bf(s[t][r]);
      }
    }

    // O += P @ V
    for (int kk = 0; kk < 2; ++kk) {
      const char* pp = (const char*)Pl + ln * HH * 2 + kk * 64 + hg * 16;
      v16bf pf = ldfrag(pp, pp + 32);
      for (int t = 0; t < 4; ++t) {
#if ASYNC_STAGE
        const char* vp = Vtile + (size_t)(t * 16 + ln) * 128 + (kk * 32 + hg * 8) * 2;
#else
        const char* vp = (const char*)Vbb + (size_t)(t * 16 + ln) * TT * 2
                         + (kb + kk * 32 + hg * 8) * 2;
#endif
        v16bf vf = ldfrag(vp, vp + 32);
        o[t] = __builtin_amdgcn_wmma_f32_16x16x32_bf16(false, pf, false, vf,
                                                       (short)0, o[t], false, false);
      }
    }
  }

  // normalize + store fp32 output [B,T,H]
  for (int t = 0; t < 4; ++t)
    for (int r = 0; r < 8; ++r) {
      const int m = r + 8 * hg;
      out[(rowbase + q0 + m) * HH + t * 16 + ln] = o[t][r] / li[r];
    }
}

// ---------------- launcher ----------------

extern "C" void kernel_launch(void* const* d_in, const int* in_sizes, int n_in,
                              void* d_out, int out_size, void* d_ws, size_t ws_size,
                              hipStream_t stream) {
  const float* x          = (const float*)d_in[0];
  const unsigned char* pm = (const unsigned char*)d_in[1];
  const float* Wq = (const float*)d_in[2];
  const float* bq = (const float*)d_in[3];
  const float* Wk = (const float*)d_in[4];
  const float* bk = (const float*)d_in[5];
  const float* Wv = (const float*)d_in[6];
  const float* bv = (const float*)d_in[7];
  float* out = (float*)d_out;

  char* ws = (char*)d_ws;
  unsigned short* Xb = (unsigned short*)ws;                                    // 16 MB
  unsigned short* Wt = (unsigned short*)(ws + (size_t)BB * TT * DD * 2);       // 192 KB
  unsigned short* Qb = (unsigned short*)((char*)Wt + (size_t)3 * HH * DD * 2); // 2 MB
  unsigned short* Kb = (unsigned short*)((char*)Qb + (size_t)BB * TT * HH * 2);
  unsigned short* Vt = (unsigned short*)((char*)Kb + (size_t)BB * TT * HH * 2);

  const int n4 = BB * TT * DD / 4;
  cvt_x<<<(n4 + 255) / 256, 256, 0, stream>>>(x, Xb, n4);
  cvt_w<<<(3 * DD * HH + 255) / 256, 256, 0, stream>>>(Wq, Wk, Wv, Wt);

  dim3 pg(BB * TT / 16, 3);
  proj<<<pg, 32, 0, stream>>>(Xb, Wt, bq, bk, bv, Qb, Kb, Vt);

  flash<<<BB * TT / 64, 128, 0, stream>>>(Qb, Kb, Vt, pm, out);
}
